// Attention_9732395892949
// MI455X (gfx1250) — matrix operand order repaired and checked by reading, not compiled, not run on hardware
//
#include <hip/hip_runtime.h>
#include <hip/hip_bf16.h>
#include <math.h>

// ---------------------------------------------------------------------------
// Attention block with per-head LoRA (q,v), tuned for MI455X (gfx1250):
//   - wave32, WMMA 16x16x32 bf16 -> f32 accumulate for every matmul
//   - flash-style online-softmax attention (never materializes NxN scores)
//   - WMMA fragments loaded as contiguous ds_load_b128 pairs
//   - row-major tiles staged with gfx1250 async global->LDS DMA (ASYNCcnt)
//   - all global accesses: uniform SGPR base + 32-bit VGPR offset
// ---------------------------------------------------------------------------

typedef __bf16 bf16_t;
typedef bf16_t v16bf __attribute__((ext_vector_type(16)));
typedef float  v8f   __attribute__((ext_vector_type(8)));
typedef unsigned int uint32;

#define BB 2
#define NN 2048
#define CC 1024
#define HH 16
#define HD 64
#define RR 16
#define NO (3 * CC)            // 3072
#define MM (BB * NN)           // 4096
#define BHN (BB * HH * NN)     // 65536
#define BHND ((size_t)BHN * HD)
#define LORA_SCALE 2.0f        // alpha / r = 32/16
#define ATTN_SCALE 0.125f      // HD^-0.5

#define LDA 40                 // stride for 32-deep (K) tiles: 80B rows, 16B-aligned
#define LDB 72                 // stride for 64-deep tiles: 144B rows, 16B-aligned

static __device__ __forceinline__ v8f v8f_zero() {
  v8f z;
#pragma unroll
  for (int i = 0; i < 8; ++i) z[i] = 0.0f;
  return z;
}

static __device__ __forceinline__ v8f wmma_bf16(v16bf a, v16bf b, v8f c) {
  return __builtin_amdgcn_wmma_f32_16x16x32_bf16(false, a, false, b,
                                                 (short)0, c, false, false);
}

// A fragment: 16x32 (MxK) from row-major LDS tile, leading dim `ld`.
static __device__ __forceinline__ v16bf load_a_frag(const bf16_t* p, int ld) {
  const int lane = threadIdx.x & 31;
  const bf16_t* r = p + (lane & 15) * ld + (lane >> 4) * 8;
  v16bf a;
  ((uint4*)&a)[0] = *(const uint4*)(r);
  ((uint4*)&a)[1] = *(const uint4*)(r + 16);
  return a;
}

// B fragment: 32x16 (KxN) from an N-MAJOR LDS tile: element (k,n) at p[n*ld+k].
static __device__ __forceinline__ v16bf load_b_frag_t(const bf16_t* p, int ld) {
  const int lane = threadIdx.x & 31;
  const bf16_t* r = p + (lane & 15) * ld + (lane >> 4) * 8;
  v16bf b;
  ((uint4*)&b)[0] = *(const uint4*)(r);
  ((uint4*)&b)[1] = *(const uint4*)(r + 16);
  return b;
}

// pack two f32 -> one u32 of bf16 pair (v_cvt_pk_bf16_f32)
static __device__ __forceinline__ uint32 pack_bf16(float f0, float f1) {
  union { bf16_t h[2]; uint32 u; } cv;
  cv.h[0] = (bf16_t)f0;
  cv.h[1] = (bf16_t)f1;
  return cv.u;
}

// gfx1250 async DMA: copy 16B from global (base + goff) into LDS byte offset.
static __device__ __forceinline__ void async_copy16(uint32 lds_off,
                                                    const void* gbase,
                                                    uint32 goff) {
  asm volatile("global_load_async_to_lds_b128 %0, %1, %2"
               :
               : "v"(lds_off), "v"(goff),
                 "s"((unsigned long long)(size_t)gbase)
               : "memory");
}
static __device__ __forceinline__ void async_wait0() {
  asm volatile("s_wait_asynccnt 0x0" ::: "memory");
}
static __device__ __forceinline__ uint32 lds_base_u32(const void* p) {
  return (uint32)(size_t)p;   // low 32 bits of generic addr == LDS offset
}

// ---------------------------------------------------------------------------
// Kernel 1: qkv = x @ w_qkv + b_qkv, scattered to bf16 [3][B][H][N][HD]
// ---------------------------------------------------------------------------
__global__ __launch_bounds__(256) void qkv_gemm_kernel(
    const float* __restrict__ x, const float* __restrict__ w,
    const float* __restrict__ bias, bf16_t* __restrict__ qkv) {
  __shared__ __align__(16) bf16_t xs[128 * LDA];   // A tile, row-major M x K
  __shared__ __align__(16) bf16_t wsb[64 * LDA];   // B tile, N-major  N x K

  const int m0 = blockIdx.x * 128;
  const int j0 = blockIdx.y * 64;
  const int tid = threadIdx.x;
  const int w8 = tid >> 5, lane = tid & 31;
  const int wr = w8 & 3, wc = w8 >> 2;
  const int half = lane >> 4, n16 = lane & 15;

  const float* xb = x + (size_t)m0 * CC;   // uniform bases; 32-bit offsets below
  const float* wb = w + j0;

  v8f acc[2][2];
#pragma unroll
  for (int i = 0; i < 2; ++i)
#pragma unroll
    for (int j = 0; j < 2; ++j) acc[i][j] = v8f_zero();

  for (int k0 = 0; k0 < CC; k0 += 32) {
#pragma unroll
    for (int it = 0; it < 8; ++it) {            // 128x32 A tile, f32->bf16 pairs
      int id = tid + it * 256;                  // 2048 pairs
      int r = id >> 4, c2 = (id & 15) * 2;
      uint32 xo = (uint32)(r * CC + k0 + c2);
      *(uint32*)(xs + r * LDA + c2) = pack_bf16(xb[xo], xb[xo + 1]);
    }
#pragma unroll
    for (int it = 0; it < 4; ++it) {            // 32x64 B tile -> N-major, K pairs
      int id = tid + it * 256;                  // 1024 pairs
      int r = (id >> 6) * 2, c = id & 63;
      uint32 wo = (uint32)((k0 + r) * NO + c);
      *(uint32*)(wsb + c * LDA + r) = pack_bf16(wb[wo], wb[wo + NO]);
    }
    if (k0 + 32 < CC) {
      __builtin_prefetch(xb + (uint32)((tid >> 1) * CC + k0 + 32));
      __builtin_prefetch(wb + (uint32)((k0 + 32 + (tid >> 3)) * NO));
    }
    __syncthreads();
    v16bf a0 = load_a_frag(xs + (wr * 32 + 0) * LDA, LDA);
    v16bf a1 = load_a_frag(xs + (wr * 32 + 16) * LDA, LDA);
    v16bf b0 = load_b_frag_t(wsb + (wc * 32 + 0) * LDA, LDA);
    v16bf b1 = load_b_frag_t(wsb + (wc * 32 + 16) * LDA, LDA);
    acc[0][0] = wmma_bf16(a0, b0, acc[0][0]);
    acc[0][1] = wmma_bf16(a0, b1, acc[0][1]);
    acc[1][0] = wmma_bf16(a1, b0, acc[1][0]);
    acc[1][1] = wmma_bf16(a1, b1, acc[1][1]);
    __syncthreads();
  }

  // Epilogue: j0 is 64-aligned and m0 is 128-aligned, so which/head/batch are
  // block constants and the scatter is linear: dst[lr*HD + lc].
  const int which = j0 >> 10;
  const int hh = (j0 >> 6) & (HH - 1);
  const int bb = m0 >> 11, nn0 = m0 & (NN - 1);
  bf16_t* dst = qkv + (size_t)which * BHND +
                (((size_t)bb * HH + hh) * NN + nn0) * HD;
  const float* bsb = bias + j0;
#pragma unroll
  for (int i = 0; i < 2; ++i)
#pragma unroll
    for (int j = 0; j < 2; ++j)
#pragma unroll
      for (int r = 0; r < 8; ++r) {
        int lr = wr * 32 + i * 16 + r + half * 8;
        int lc = wc * 32 + j * 16 + n16;
        float vv = acc[i][j][r] + bsb[lc];
        dst[(uint32)(lr * HD + lc)] = (bf16_t)vv;
      }
}

// ---------------------------------------------------------------------------
// Kernel 2: in-place LoRA  buf += ((buf @ A) @ Bm) * LORA_SCALE
// ---------------------------------------------------------------------------
__global__ __launch_bounds__(256) void lora_kernel(
    bf16_t* __restrict__ buf, const float* __restrict__ A,
    const float* __restrict__ Bm) {
  __shared__ __align__(16) bf16_t qs[128 * LDB];   // rows, row-major (M x K=64)
  __shared__ __align__(16) bf16_t aSt[16 * LDB];   // A  N-major: [n=16][k=64]
  __shared__ __align__(16) bf16_t bSt[64 * LDA];   // Bm N-major: [n=64][k=32 pad]
  __shared__ __align__(16) bf16_t tS[8 * 16 * LDA];

  const int row0 = blockIdx.x * 128;
  const int tid = threadIdx.x, w8 = tid >> 5, lane = tid & 31;
  const int half = lane >> 4, n16 = lane & 15;

  bf16_t* bufb = buf + (size_t)row0 * HD;
  const uint32 qsb = lds_base_u32(qs);
#pragma unroll
  for (int it = 0; it < 4; ++it) {             // 128x64 bf16 tile, async DMA
    int id = tid + it * 256;                   // 1024 x 16B chunks
    int r = id >> 3, c = id & 7;
    async_copy16(qsb + r * (LDB * 2) + c * 16, bufb, (uint32)(r * 128 + c * 16));
  }
#pragma unroll
  for (int it = 0; it < 4; ++it) {             // A (64x16) -> N-major [16][64]
    int id = tid + it * 256;
    int k = id >> 4, n = id & 15;
    aSt[n * LDB + k] = (bf16_t)A[id];
  }
#pragma unroll
  for (int it = 0; it < 2; ++it) {             // Bm (16x64) -> N-major, K pairs
    int id = tid + it * 256;                   // 512 pairs
    int r = (id >> 6) * 2, c = id & 63;
    *(uint32*)(bSt + c * LDA + r) = pack_bf16(Bm[r * HD + c], Bm[(r + 1) * HD + c]);
    *(uint32*)(bSt + c * LDA + 16 + r) = 0u;   // zero pad K=16..31
  }
  async_wait0();
  __syncthreads();

  v16bf qA0 = load_a_frag(qs + w8 * 16 * LDB + 0, LDB);
  v16bf qA1 = load_a_frag(qs + w8 * 16 * LDB + 32, LDB);
  v16bf aB0 = load_b_frag_t(aSt + 0, LDB);
  v16bf aB1 = load_b_frag_t(aSt + 32, LDB);

  v8f t = v8f_zero();                          // tmp = rows @ A  (16x16)
  t = wmma_bf16(qA0, aB0, t);
  t = wmma_bf16(qA1, aB1, t);

  bf16_t* tw = tS + w8 * 16 * LDA;             // stage tmp as A-frag, K pad
#pragma unroll
  for (int r = 0; r < 8; ++r) {
    int rr = r + half * 8;
    tw[rr * LDA + n16] = (bf16_t)t[r];
    tw[rr * LDA + 16 + n16] = (bf16_t)0.0f;
  }
  v16bf tA = load_a_frag(tw, LDA);

#pragma unroll
  for (int j = 0; j < 4; ++j) {                // delta = tmp @ Bm (16x64)
    v8f d = v8f_zero();
    d = wmma_bf16(tA, load_b_frag_t(bSt + (j * 16) * LDA, LDA), d);
#pragma unroll
    for (int r = 0; r < 8; ++r) {
      int rr = r + half * 8;
      int gc = j * 16 + n16;
      float o = (float)qs[(w8 * 16 + rr) * LDB + gc] + LORA_SCALE * d[r];
      bufb[(uint32)((w8 * 16 + rr) * HD + gc)] = (bf16_t)o;
    }
  }
}

// ---------------------------------------------------------------------------
// Kernel 3: flash attention per (b,h). 128 q rows / WG, 64-wide K/V tiles.
// ---------------------------------------------------------------------------
__global__ __launch_bounds__(256) void attn_kernel(
    const bf16_t* __restrict__ q, const bf16_t* __restrict__ k,
    const bf16_t* __restrict__ v, bf16_t* __restrict__ ao) {
  __shared__ __align__(16) bf16_t qs[128 * LDB];
  __shared__ __align__(16) bf16_t ks[64 * LDB];    // [kv][d]
  __shared__ __align__(16) bf16_t vsT[64 * LDB];   // [d][kv]
  __shared__ __align__(16) bf16_t ps[8 * 16 * LDB];

  const int bh = blockIdx.x;
  const int n0 = blockIdx.y * 128;
  const int tid = threadIdx.x, w8 = tid >> 5, lane = tid & 31;
  const int half = lane >> 4, n16 = lane & 15;
  const size_t base = (size_t)bh * NN * HD;
  const bf16_t* qg = q + base + (size_t)n0 * HD;
  const bf16_t* kg = k + base;
  const bf16_t* vg = v + base;

  const uint32 qsb = lds_base_u32(qs);
  const uint32 ksb = lds_base_u32(ks);
#pragma unroll
  for (int it = 0; it < 4; ++it) {             // q tile: async DMA, 1024 chunks
    int id = tid + it * 256;
    int r = id >> 3, c = id & 7;
    async_copy16(qsb + r * (LDB * 2) + c * 16, qg, (uint32)(r * 128 + c * 16));
  }
  async_wait0();
  __syncthreads();
  v16bf qA0 = load_a_frag(qs + w8 * 16 * LDB + 0, LDB);
  v16bf qA1 = load_a_frag(qs + w8 * 16 * LDB + 32, LDB);

  float mrow[8], lrow[8];
  v8f oacc[4];
#pragma unroll
  for (int r = 0; r < 8; ++r) { mrow[r] = -1e30f; lrow[r] = 0.0f; }
#pragma unroll
  for (int j = 0; j < 4; ++j) oacc[j] = v8f_zero();

  for (int kv0 = 0; kv0 < NN; kv0 += 64) {
#pragma unroll
    for (int it = 0; it < 2; ++it) {           // k tile: async DMA [kv][d]
      int id = tid + it * 256;                 // 512 chunks
      int r = id >> 3, c = id & 7;
      async_copy16(ksb + r * (LDB * 2) + c * 16, kg,
                   (uint32)((kv0 + r) * 128 + c * 16));
    }
#pragma unroll
    for (int it = 0; it < 8; ++it) {           // v tile: transpose, kv pairs
      int id = tid + it * 256;                 // 2048 pairs
      int kvr = (id >> 6) * 2, d = id & 63;
      uint32 vo = (uint32)((kv0 + kvr) * HD + d);
      *(uint32*)(vsT + d * LDB + kvr) = pack_bf16((float)vg[vo], (float)vg[vo + HD]);
    }
    if (kv0 + 64 < NN) {
      __builtin_prefetch(kg + (uint32)((kv0 + 64) * HD) + tid * 16);
      __builtin_prefetch(vg + (uint32)((kv0 + 64) * HD) + tid * 16);
    }
    async_wait0();
    __syncthreads();

    v8f s[4];                                   // S = q @ k^T (16 x 64)
#pragma unroll
    for (int j = 0; j < 4; ++j) {
      v8f z = v8f_zero();
      z = wmma_bf16(qA0, load_b_frag_t(ks + (j * 16) * LDB + 0, LDB), z);
      z = wmma_bf16(qA1, load_b_frag_t(ks + (j * 16) * LDB + 32, LDB), z);
      s[j] = z;
    }

    float nm[8], al[8], rs[8];
#pragma unroll
    for (int r = 0; r < 8; ++r) {
      float pm = -1e30f;
#pragma unroll
      for (int j = 0; j < 4; ++j) {
        s[j][r] *= ATTN_SCALE;
        pm = fmaxf(pm, s[j][r]);
      }
#pragma unroll
      for (int off = 8; off > 0; off >>= 1)     // reduce within 16-lane half
        pm = fmaxf(pm, __shfl_xor(pm, off, 32));
      nm[r] = fmaxf(mrow[r], pm);
      al[r] = __expf(mrow[r] - nm[r]);
      rs[r] = 0.0f;
    }

    bf16_t* pw = ps + w8 * 16 * LDB;            // P = exp(S - m), staged bf16
#pragma unroll
    for (int j = 0; j < 4; ++j)
#pragma unroll
      for (int r = 0; r < 8; ++r) {
        float p = __expf(s[j][r] - nm[r]);
        rs[r] += p;
        pw[(r + half * 8) * LDB + j * 16 + n16] = (bf16_t)p;
      }
#pragma unroll
    for (int r = 0; r < 8; ++r) {
#pragma unroll
      for (int off = 8; off > 0; off >>= 1) rs[r] += __shfl_xor(rs[r], off, 32);
      lrow[r] = lrow[r] * al[r] + rs[r];
      mrow[r] = nm[r];
    }
#pragma unroll
    for (int j = 0; j < 4; ++j)
#pragma unroll
      for (int r = 0; r < 8; ++r) oacc[j][r] *= al[r];

    v16bf pA0 = load_a_frag(pw + 0, LDB);       // O += P @ V
    v16bf pA1 = load_a_frag(pw + 32, LDB);
#pragma unroll
    for (int j = 0; j < 4; ++j) {
      oacc[j] = wmma_bf16(pA0, load_b_frag_t(vsT + (j * 16) * LDB + 0, LDB),
                          oacc[j]);
      oacc[j] = wmma_bf16(pA1, load_b_frag_t(vsT + (j * 16) * LDB + 32, LDB),
                          oacc[j]);
    }
    __syncthreads();
  }

  const int b = bh >> 4, h = bh & (HH - 1);     // scatter to [B][N][C] bf16
  bf16_t* aob = ao + ((size_t)(b * NN + n0)) * CC + h * HD;
#pragma unroll
  for (int j = 0; j < 4; ++j)
#pragma unroll
    for (int r = 0; r < 8; ++r) {
      int rr = r + half * 8;
      float o = oacc[j][r] / lrow[r];
      aob[(uint32)((w8 * 16 + rr) * CC + j * 16 + n16)] = (bf16_t)o;
    }
}

// ---------------------------------------------------------------------------
// Kernel 4: out = attn_out @ w_proj + b_proj  (f32 output)
// ---------------------------------------------------------------------------
__global__ __launch_bounds__(256) void proj_gemm_kernel(
    const bf16_t* __restrict__ a, const float* __restrict__ w,
    const float* __restrict__ bias, float* __restrict__ out) {
  __shared__ __align__(16) bf16_t xs[128 * LDA];   // A tile, row-major
  __shared__ __align__(16) bf16_t wsb[64 * LDA];   // B tile, N-major

  const int m0 = blockIdx.x * 128;
  const int j0 = blockIdx.y * 64;
  const int tid = threadIdx.x;
  const int w8 = tid >> 5, lane = tid & 31;
  const int wr = w8 & 3, wc = w8 >> 2;
  const int half = lane >> 4, n16 = lane & 15;
  const uint32 xsb = lds_base_u32(xs);
  const bf16_t* arow = a + (size_t)m0 * CC;
  const float* wb = w + j0;

  v8f acc[2][2];
#pragma unroll
  for (int i = 0; i < 2; ++i)
#pragma unroll
    for (int j = 0; j < 2; ++j) acc[i][j] = v8f_zero();

  for (int k0 = 0; k0 < CC; k0 += 32) {
#pragma unroll
    for (int it = 0; it < 2; ++it) {            // 128x32 bf16 A tile: async DMA
      int id = tid + it * 256;                  // 512 x 16B chunks
      int r = id >> 2, c = id & 3;
      async_copy16(xsb + r * (LDA * 2) + c * 16, arow,
                   (uint32)(r * (CC * 2) + k0 * 2 + c * 16));
    }
#pragma unroll
    for (int it = 0; it < 4; ++it) {            // 32x64 f32->bf16 -> N-major pairs
      int id = tid + it * 256;                  // 1024 pairs
      int r = (id >> 6) * 2, c = id & 63;
      uint32 wo = (uint32)((k0 + r) * CC + c);
      *(uint32*)(wsb + c * LDA + r) = pack_bf16(wb[wo], wb[wo + CC]);
    }
    if (k0 + 32 < CC) {
      __builtin_prefetch(arow + (uint32)((tid >> 1) * CC + k0 + 32));
      __builtin_prefetch(wb + (uint32)((k0 + 32 + (tid >> 3)) * CC));
    }
    async_wait0();
    __syncthreads();
    v16bf a0 = load_a_frag(xs + (wr * 32 + 0) * LDA, LDA);
    v16bf a1 = load_a_frag(xs + (wr * 32 + 16) * LDA, LDA);
    v16bf b0 = load_b_frag_t(wsb + (wc * 32 + 0) * LDA, LDA);
    v16bf b1 = load_b_frag_t(wsb + (wc * 32 + 16) * LDA, LDA);
    acc[0][0] = wmma_bf16(a0, b0, acc[0][0]);
    acc[0][1] = wmma_bf16(a0, b1, acc[0][1]);
    acc[1][0] = wmma_bf16(a1, b0, acc[1][0]);
    acc[1][1] = wmma_bf16(a1, b1, acc[1][1]);
    __syncthreads();
  }

  float* outb = out + (size_t)m0 * CC + j0;
  const float* bsb = bias + j0;
#pragma unroll
  for (int i = 0; i < 2; ++i)
#pragma unroll
    for (int j = 0; j < 2; ++j)
#pragma unroll
      for (int r = 0; r < 8; ++r) {
        int lr = wr * 32 + i * 16 + r + half * 8;
        int lc = wc * 32 + j * 16 + n16;
        outb[(uint32)(lr * CC + lc)] = acc[i][j][r] + bsb[lc];
      }
}

// ---------------------------------------------------------------------------
extern "C" void kernel_launch(void* const* d_in, const int* in_sizes, int n_in,
                              void* d_out, int out_size, void* d_ws,
                              size_t ws_size, hipStream_t stream) {
  const float* x      = (const float*)d_in[0];
  const float* w_qkv  = (const float*)d_in[1];
  const float* b_qkv  = (const float*)d_in[2];
  const float* a_q    = (const float*)d_in[3];
  const float* b_q    = (const float*)d_in[4];
  const float* a_v    = (const float*)d_in[5];
  const float* b_v    = (const float*)d_in[6];
  const float* w_proj = (const float*)d_in[7];
  const float* b_proj = (const float*)d_in[8];
  float* out = (float*)d_out;

  // Workspace layout (bf16): q|k|v [B][H][N][HD] then attn-out [B][N][C]
  bf16_t* qkv  = (bf16_t*)d_ws;
  bf16_t* qb   = qkv;
  bf16_t* kb   = qkv + BHND;
  bf16_t* vb   = qkv + 2 * BHND;
  bf16_t* abuf = qkv + 3 * BHND;

  qkv_gemm_kernel<<<dim3(MM / 128, NO / 64), 256, 0, stream>>>(x, w_qkv, b_qkv,
                                                               qkv);
  lora_kernel<<<dim3(BHN / 128), 256, 0, stream>>>(qb, a_q, b_q);
  lora_kernel<<<dim3(BHN / 128), 256, 0, stream>>>(vb, a_v, b_v);
  attn_kernel<<<dim3(BB * HH, NN / 128), 256, 0, stream>>>(qb, kb, vb, abuf);
  proj_gemm_kernel<<<dim3(MM / 128, CC / 64), 256, 0, stream>>>(abuf, w_proj,
                                                                b_proj, out);
}
